// SimpleGATLayer_30382598652183
// MI455X (gfx1250) — compile-verified, matrix operand order checked by reading, not run on hardware
//
#include <hip/hip_runtime.h>

typedef __attribute__((ext_vector_type(2)))  float   v2f;
typedef __attribute__((ext_vector_type(8)))  float   v8f;
typedef __attribute__((ext_vector_type(8)))  __bf16  v8bf;
typedef __attribute__((ext_vector_type(16))) __bf16  v16bf;

#define NN     8192
#define FD     256
#define ALPHA  0.2f
#define NEGV   (-9.0e15f)

__device__ __forceinline__ float lrelu(float x) { return x > 0.f ? x : ALPHA * x; }

__device__ __forceinline__ float wave_sum(float x) {
  for (int off = 16; off; off >>= 1) x += __shfl_xor(x, off, 32);
  return x;
}

// -------------------------------------------------------------------------
// Kernel 1: Wh = h @ W_w + W_b  (f32 WMMA 16x16x4).
// Writes Wh (f32, row-major, for s1/s2) and WhTp: bf16 Wh pre-packed in
// B-fragment order  [kChunk(=row/32)][col][k%32]  so the big GEMM's B loads
// are dense, coalesced 16B-per-lane runs.
// grid (512, 4) x 128 threads; wave w covers col tile blockIdx.y*4 + w.
// -------------------------------------------------------------------------
__global__ __launch_bounds__(128) void gat_wh_kernel(
    const float* __restrict__ h, const float* __restrict__ Ww,
    const float* __restrict__ Wb, float* __restrict__ Wh,
    __bf16* __restrict__ WhTp)
{
  const int wave = threadIdx.x >> 5;
  const int lane = threadIdx.x & 31;
  const int half = lane >> 4;
  const int lm   = lane & 15;
  const int i0   = blockIdx.x * 16;
  const int n0   = (blockIdx.y * 4 + wave) * 16;
  const int row  = i0 + lm;
  const int col  = n0 + lm;

  v8f c = {};
  const float* __restrict__ hrow = h + (size_t)row * FD;

#if __has_builtin(__builtin_amdgcn_wmma_f32_16x16x4_f32)
  // A 16x4 f32: VGPR0 = K0 (lanes0-15) / K2 (lanes16-31); VGPR1 = K1 / K3
  for (int k0 = 0; k0 < FD; k0 += 4) {
    const int ka = k0 + 2 * half;
    v2f a, b;
    a.x = hrow[ka];
    a.y = hrow[ka + 1];
    b.x = Ww[(size_t)ka * FD + col];
    b.y = Ww[(size_t)(ka + 1) * FD + col];
    c = __builtin_amdgcn_wmma_f32_16x16x4_f32(false, a, false, b,
                                              (short)0, c, false, false);
  }
#else
#warning "wmma_f32_16x16x4_f32 builtin unavailable: using bf16 WMMA fallback"
  for (int k0 = 0; k0 < FD; k0 += 32) {
    v16bf a, b;
    for (int e = 0; e < 16; ++e) {
      const int k = k0 + ((e < 8) ? (8 * half + e) : (16 + 8 * half + (e - 8)));
      a[e] = (__bf16)hrow[k];
      b[e] = (__bf16)Ww[(size_t)k * FD + col];
    }
    c = __builtin_amdgcn_wmma_f32_16x16x32_bf16(false, a, false, b,
                                                (short)0, c, false, false);
  }
#endif

  const float bb = Wb[col];
  for (int v = 0; v < 8; ++v) {
    const int r = i0 + v + 8 * half;          // C layout: M = v (+8 hi half)
    const float val = c[v] + bb;
    Wh[(size_t)r * FD + col] = val;
    // packed B layout: chunk = r>>5, within-chunk k = r&31
    WhTp[(size_t)(r >> 5) * (FD * 32) + col * 32 + (r & 31)] = (__bf16)val;
  }
}

// -------------------------------------------------------------------------
// Kernel 2: s1[i] = Wh[i,:]·a1 ; s2[i] = Wh[i,:]·a2  (wave per row, f32)
// -------------------------------------------------------------------------
__global__ __launch_bounds__(256) void gat_s_kernel(
    const float* __restrict__ Wh, const float* __restrict__ a,
    float* __restrict__ s1, float* __restrict__ s2)
{
  const int wave = threadIdx.x >> 5;
  const int lane = threadIdx.x & 31;
  const int r    = blockIdx.x * 8 + wave;
  const float* __restrict__ wr = Wh + (size_t)r * FD;
  float acc1 = 0.f, acc2 = 0.f;
  for (int q = 0; q < FD; q += 32) {
    const float w = wr[q + lane];
    acc1 += w * a[q + lane];
    acc2 += w * a[FD + q + lane];
  }
  acc1 = wave_sum(acc1);
  acc2 = wave_sum(acc2);
  if (lane == 0) { s1[r] = acc1; s2[r] = acc2; }
}

// -------------------------------------------------------------------------
// Kernel 3: s2max = max_j s2[j]  (single block; gives the global softmax
// shift bound m̂_i = lrelu(s1_i + s2max) — no adj pass needed for the max)
// -------------------------------------------------------------------------
__global__ __launch_bounds__(256) void gat_s2max_kernel(
    const float* __restrict__ s2, float* __restrict__ s2max)
{
  __shared__ float red[256];
  const int t = threadIdx.x;
  float m = NEGV;
  for (int j = t; j < NN; j += 256) m = fmaxf(m, s2[j]);
  red[t] = m;
  __syncthreads();
  for (int s = 128; s; s >>= 1) {
    if (t < s) red[t] = fmaxf(red[t], red[t + s]);
    __syncthreads();
  }
  if (t == 0) s2max[0] = red[0];
}

// -------------------------------------------------------------------------
// Kernel 4 (fused, single pass over adj):
//   p̃_ij = adj ? exp(lrelu(s1_i+s2_j) - m̂_i) : 0   (bf16)
//   numerator  N = P̃ @ Wh     via v_wmma_f32_16x16x32_bf16
//   denominator l_i = Σ_j p̃   in f32 (from the same bf16-rounded values)
//   out = N / l               (exact softmax by shift invariance)
// Block = 32 rows x 256 cols; 8 waves own 32 cols each (2 N-tiles),
// 2 M-tiles -> 4 WMMAs per wave per 32-wide K chunk.
// -------------------------------------------------------------------------
__global__ __launch_bounds__(256) void gat_fused_kernel(
    const int* __restrict__ adj, const float* __restrict__ s1,
    const float* __restrict__ s2, const float* __restrict__ s2max,
    const __bf16* __restrict__ WhTp, float* __restrict__ out)
{
  __shared__ __align__(16) __bf16 pt[32 * 32];   // P tile, 2 KB
  __shared__ float ls[32];                       // per-row denominators

  const int t    = threadIdx.x;
  const int wave = t >> 5;
  const int lane = t & 31;
  const int half = lane >> 4;
  const int lm   = lane & 15;
  const int i0   = blockIdx.x * 32;
  const int n0   = wave * 32;                    // this wave's output cols

  // This thread fills P rows wave+{0,8,16,24} at column = lane.
  const float S2M = s2max[0];
  float s1v[4], mh[4], lacc[4];
  size_t ab[4];
  for (int q = 0; q < 4; ++q) {
    const int gi = i0 + wave + 8 * q;
    s1v[q]  = s1[gi];
    mh[q]   = lrelu(s1v[q] + S2M);               // fixed safe shift
    lacc[q] = 0.f;
    ab[q]   = (size_t)gi * NN;
  }

  // B fragment base pointers (packed layout: [chunk][col][32])
  const __bf16* __restrict__ bp0 = WhTp + (size_t)(n0 + lm) * 32 + 8 * half;
  const __bf16* __restrict__ bp1 = WhTp + (size_t)(n0 + 16 + lm) * 32 + 8 * half;

  v8f acc00 = {}, acc01 = {}, acc10 = {}, acc11 = {};

  for (int j0 = 0; j0 < NN; j0 += 32) {
    const int j = j0 + lane;
    for (int q = 0; q < 4; ++q) {
      float p = 0.f;
      if (adj[ab[q] + j] > 0)
        p = __expf(lrelu(s1v[q] + s2[j]) - mh[q]);
      const __bf16 pb = (__bf16)p;
      lacc[q] += (float)pb;                      // denominator from rounded p
      pt[(wave + 8 * q) * 32 + lane] = pb;
      if (j0 + 32 < NN)                          // global_prefetch_b8
        __builtin_prefetch(&adj[ab[q] + j + 32], 0, 1);
    }
    __syncthreads();

    // A fragments (shared by all 8 waves):
    // elems 0..7 -> K = 8*half+0..7 ; elems 8..15 -> K = 16+8*half+0..7
    const v8bf a0lo = *(const v8bf*)&pt[lm * 32 + 8 * half];
    const v8bf a0hi = *(const v8bf*)&pt[lm * 32 + 16 + 8 * half];
    const v16bf A0 = __builtin_shufflevector(a0lo, a0hi,
        0, 1, 2, 3, 4, 5, 6, 7, 8, 9, 10, 11, 12, 13, 14, 15);
    const v8bf a1lo = *(const v8bf*)&pt[(16 + lm) * 32 + 8 * half];
    const v8bf a1hi = *(const v8bf*)&pt[(16 + lm) * 32 + 16 + 8 * half];
    const v16bf A1 = __builtin_shufflevector(a1lo, a1hi,
        0, 1, 2, 3, 4, 5, 6, 7, 8, 9, 10, 11, 12, 13, 14, 15);

    // B fragments: dense coalesced reads from packed chunk (j0>>5)
    const size_t cb = (size_t)(j0 >> 5) * (FD * 32);
    const v8bf b0lo = *(const v8bf*)(bp0 + cb);
    const v8bf b0hi = *(const v8bf*)(bp0 + cb + 16);
    const v16bf B0 = __builtin_shufflevector(b0lo, b0hi,
        0, 1, 2, 3, 4, 5, 6, 7, 8, 9, 10, 11, 12, 13, 14, 15);
    const v8bf b1lo = *(const v8bf*)(bp1 + cb);
    const v8bf b1hi = *(const v8bf*)(bp1 + cb + 16);
    const v16bf B1 = __builtin_shufflevector(b1lo, b1hi,
        0, 1, 2, 3, 4, 5, 6, 7, 8, 9, 10, 11, 12, 13, 14, 15);

    acc00 = __builtin_amdgcn_wmma_f32_16x16x32_bf16(false, A0, false, B0,
                                                    (short)0, acc00, false, false);
    acc01 = __builtin_amdgcn_wmma_f32_16x16x32_bf16(false, A0, false, B1,
                                                    (short)0, acc01, false, false);
    acc10 = __builtin_amdgcn_wmma_f32_16x16x32_bf16(false, A1, false, B0,
                                                    (short)0, acc10, false, false);
    acc11 = __builtin_amdgcn_wmma_f32_16x16x32_bf16(false, A1, false, B1,
                                                    (short)0, acc11, false, false);
    __syncthreads();
  }

  // Per-row denominators: wave-reduce this thread's 4 rows, publish to LDS.
  for (int q = 0; q < 4; ++q) {
    const float l = wave_sum(lacc[q]);
    if (lane == 0) ls[wave + 8 * q] = l;
  }
  __syncthreads();

  // Divide and store: acc00/acc01 cover local rows v+8*half,
  // acc10/acc11 cover local rows 16+v+8*half; col = lane%16 within tile.
  for (int v = 0; v < 8; ++v) {
    const int r0 = v + 8 * half;
    const int r1 = 16 + r0;
    const float il0 = 1.f / ls[r0];
    const float il1 = 1.f / ls[r1];
    float* o0 = out + (size_t)(i0 + r0) * FD;
    float* o1 = out + (size_t)(i0 + r1) * FD;
    o0[n0 + lm]      = acc00[v] * il0;
    o0[n0 + 16 + lm] = acc01[v] * il0;
    o1[n0 + lm]      = acc10[v] * il1;
    o1[n0 + 16 + lm] = acc11[v] * il1;
  }
}

// -------------------------------------------------------------------------
extern "C" void kernel_launch(void* const* d_in, const int* in_sizes, int n_in,
                              void* d_out, int out_size, void* d_ws, size_t ws_size,
                              hipStream_t stream) {
  (void)in_sizes; (void)n_in; (void)out_size; (void)ws_size;
  const float* h   = (const float*)d_in[0];
  const int*   adj = (const int*)d_in[1];
  const float* Ww  = (const float*)d_in[2];
  const float* Wb  = (const float*)d_in[3];
  const float* a   = (const float*)d_in[4];

  char*   ws   = (char*)d_ws;
  float*  Wh   = (float*)ws;                            // 8 MB
  __bf16* WhTp = (__bf16*)(ws + 8388608);               // 4 MB (packed B)
  float*  s1   = (float*)(ws + 8388608 + 4194304);      // 32 KB
  float*  s2   = s1 + NN;                               // 32 KB
  float*  s2m  = s2 + NN;                               // 4 B
  float*  out  = (float*)d_out;

  gat_wh_kernel    <<<dim3(NN / 16, 4), 128, 0, stream>>>(h, Ww, Wb, Wh, WhTp);
  gat_s_kernel     <<<NN / 8, 256, 0, stream>>>(Wh, a, s1, s2);
  gat_s2max_kernel <<<1, 256, 0, stream>>>(s2, s2m);
  gat_fused_kernel <<<NN / 32, 256, 0, stream>>>(adj, s1, s2, s2m, WhTp, out);
}
